// ZipUNet_88708254532284
// MI455X (gfx1250) — matrix operand order repaired and checked by reading, not compiled, not run on hardware
//
#include <hip/hip_runtime.h>
#include <hip/hip_bf16.h>

typedef __attribute__((ext_vector_type(16))) __bf16       v16bf;
typedef __attribute__((ext_vector_type(8)))  float        v8f;
typedef __attribute__((ext_vector_type(8)))  unsigned int u32x8;
typedef __attribute__((ext_vector_type(4)))  unsigned int u32x4;
typedef __attribute__((ext_vector_type(8)))  int          i32x8;
typedef __attribute__((ext_vector_type(4)))  int          i32x4;

#define BM 64
#define BN 128
#define BK 32
#define AKP 40    // A tile LDS row stride in floats (32 data + 8 TDM pad dwords)
#define BNP 136   // B tile LDS row stride in halfs
#define NB 8      // batch

// f32 -> bf16 with round-to-nearest-even
__device__ __forceinline__ unsigned short f2bf(float f) {
  unsigned int u = __builtin_bit_cast(unsigned int, f);
  u += 0x7FFFu + ((u >> 16) & 1u);
  return (unsigned short)(u >> 16);
}

// ---------------------------------------------------------------------------
// TDM: async 2D tile load  W[m0.., k0..k0+32) -> LDS (f32, row stride 40 dw).
// Descriptor built per CDNA5 ISA ch.8 (D# group0/group1), 2D -> groups 2/3 zero.
// pad_interval=4 (32 dwords) + pad_amount=7 (8 dwords) => LDS row stride 40 dw.
// tensor_dim relative to tile start => rows >= OC-m0 read as zero (OOB fill).
// ---------------------------------------------------------------------------
__device__ __forceinline__ void tdm_load_A(const float* Wt, unsigned lds_addr,
                                           int m0, int k0, int OC, int K)
{
#if __has_builtin(__builtin_amdgcn_tensor_load_to_lds)
  unsigned long long ga =
      (unsigned long long)(const void*)(Wt + (size_t)m0 * K + k0);
  unsigned td0 = (unsigned)(K - k0);    // remaining elems in a row
  unsigned td1 = (unsigned)(OC - m0);   // remaining rows
  unsigned long long s0 = (unsigned long long)K;  // row stride (elements)

  u32x4 g0;
  g0[0] = 1u;                                   // count=1, no gather
  g0[1] = lds_addr;                             // LDS byte address
  g0[2] = (unsigned)ga;                         // global_addr[31:0]
  g0[3] = (unsigned)((ga >> 32) & 0x01FFFFFFull) | (2u << 30);  // addr hi|type=2

  u32x8 g1u;
  g1u[0] = (2u << 16)            // data_size = 4B
         | (1u << 20)            // pad_enable
         | (4u << 22)            // pad_interval: 32 dwords
         | (7u << 25);           // pad_amount : 8 dwords
  g1u[1] = (td0 & 0xFFFFu) << 16;                       // dim0[15:0]
  g1u[2] = ((td0 >> 16) & 0xFFFFu) | ((td1 & 0xFFFFu) << 16);
  g1u[3] = ((td1 >> 16) & 0xFFFFu) | ((unsigned)BK << 16);  // tile_dim0=32
  g1u[4] = (unsigned)BM;                                // tile_dim1=64, dim2=0
  g1u[5] = (unsigned)(s0 & 0xFFFFFFFFull);              // dim0_stride lo
  g1u[6] = (unsigned)((s0 >> 32) & 0xFFFFull);          // dim0_stride hi
  g1u[7] = 0u;
  i32x8 g1 = __builtin_bit_cast(i32x8, g1u);
  i32x4 gz = {};
#if defined(__clang_major__) && __clang_major__ >= 23
  i32x8 g4 = {};
  __builtin_amdgcn_tensor_load_to_lds(g0, g1, gz, gz, g4, 0);
#else
  __builtin_amdgcn_tensor_load_to_lds(g0, g1, gz, gz, 0);
#endif
#else
  (void)Wt; (void)lds_addr; (void)m0; (void)k0; (void)OC; (void)K;
#endif
}

// ---------------------------------------------------------------------------
// Implicit-GEMM conv / transposed-conv via v_wmma_f32_16x16x32_bf16.
//   D[OC, B*OH*OW] = W[OC, IC*16] x Patch[IC*16, B*OH*OW]
//   transposed==0 : conv k4 s2 p1   (iy = 2*oy-1+ky)
//   transposed==1 : convT k4 s2 p1  (iy = (oy+ky-2)/2, even parity; weights
//                                    pre-flipped by the reference)
//   act (input, fused in gather): 0=none 1=leaky(0.2) 2=relu
//   ep : 0=store, 1=+bias then tanh
//   Tile: 64x128 block, 8 waves 4Mx2N, wave = 16x64 -> 4 WMMA / chunk, A reused.
// ---------------------------------------------------------------------------
__global__ __launch_bounds__(256)
void conv_wmma(const float* __restrict__ Wt, const float* __restrict__ X,
               float* __restrict__ Y, const float* __restrict__ bias,
               int OC, int IC, int IH, int IW, int OH, int OW,
               int xct, int xc0, int yct, int yc0,
               int act, int transposed, int ep)
{
  __shared__ float          Afs[BM][AKP];  // A tile (f32, TDM-filled, pad 40)
  __shared__ unsigned short Bs[BK][BNP];   // B tile (bf16 im2col patches)

  const int K    = IC * 16;
  const int OHW  = OH * OW;
  const int Ntot = NB * OHW;
  const int m0   = blockIdx.y * BM;
  const int n0   = blockIdx.x * BN;

  const int lane = threadIdx.x & 31;
  const int wave = threadIdx.x >> 5;
  const int wm   = wave >> 1;   // 0..3 : 16-row slab
  const int wn   = wave & 1;    // 0..1 : 64-col slab
  const int ml   = lane & 15;
  const int hi   = lane >> 4;   // lane half

  const unsigned ldsA = (unsigned)(unsigned long long)(const void*)&Afs[0][0];

  v8f acc[4] = {};

  const bool tdm_ok =
#if __has_builtin(__builtin_amdgcn_tensor_load_to_lds)
      true;
#else
      false;
#endif

  for (int k0 = 0; k0 < K; k0 += BK) {
    __syncthreads();

    // ---- A tile: issue async TDM from wave 0 (EXEC ignored by TDM) -------
    if (tdm_ok) {
      if (wave == 0) tdm_load_A(Wt, ldsA, m0, k0, OC, K);
    } else {
      for (int i = threadIdx.x; i < BM * BK; i += 256) {
        int m = i >> 5, k = i & (BK - 1);
        int oc = m0 + m, kk = k0 + k;
        Afs[m][k] = (oc < OC && kk < K) ? Wt[oc * K + kk] : 0.0f;
      }
    }

    // ---- B tile: im2col gather with fused pre-activation (overlaps TDM) --
    for (int i = threadIdx.x; i < BK * BN; i += 256) {
      int k = i >> 7, n = i & (BN - 1);
      int kk = k0 + k, nn = n0 + n;
      float v = 0.0f;
      if (kk < K && nn < Ntot) {
        int ic = kk >> 4, t = kk & 15, ky = t >> 2, kx = t & 3;
        int b = nn / OHW, r = nn - b * OHW;
        int oy = r / OW, ox = r - oy * OW;
        int iy, ix; bool ok;
        if (!transposed) {
          iy = oy * 2 - 1 + ky;  ix = ox * 2 - 1 + kx;
          ok = (iy >= 0) & (iy < IH) & (ix >= 0) & (ix < IW);
        } else {
          int dy = oy + ky - 2, dx = ox + kx - 2;
          ok = (dy >= 0) & (dx >= 0) & !(dy & 1) & !(dx & 1);
          iy = dy >> 1;  ix = dx >> 1;
          ok = ok & (iy < IH) & (ix < IW);
        }
        if (ok) {
          v = X[((b * xct) + xc0 + ic) * IH * IW + iy * IW + ix];
          if      (act == 1) v = v > 0.0f ? v : 0.2f * v;
          else if (act == 2) v = v > 0.0f ? v : 0.0f;
        }
      }
      Bs[k][n] = f2bf(v);
    }

    if (tdm_ok && wave == 0)
      __builtin_amdgcn_s_wait_tensorcnt(0);   // tile landed in LDS
    __syncthreads();

    // ---- fragments per ISA 7.12.2 layouts; 4 WMMAs reusing one A ---------
    u32x8 afr;
    const int mr = wm * 16 + ml;
    const int nc = wn * 64 + ml;
#pragma unroll
    for (int v = 0; v < 8; ++v) {
      int ka = (v < 4) ? (2 * v + hi * 8) : (16 + 2 * (v - 4) + hi * 8);
      afr[v] = (unsigned)f2bf(Afs[mr][ka]) |
               ((unsigned)f2bf(Afs[mr][ka + 1]) << 16);
    }
    v16bf Af = __builtin_bit_cast(v16bf, afr);
#pragma unroll
    for (int h = 0; h < 4; ++h) {
      u32x8 bfr;
#pragma unroll
      for (int v = 0; v < 8; ++v) {
        int kb = (v < 4) ? (2 * v + hi * 8) : (16 + 2 * (v - 4) + hi * 8);
        int cc = nc + h * 16;
        bfr[v] = (unsigned)Bs[kb][cc] | ((unsigned)Bs[kb + 1][cc] << 16);
      }
      v16bf Bf = __builtin_bit_cast(v16bf, bfr);
      acc[h] = __builtin_amdgcn_wmma_f32_16x16x32_bf16(false, Af, false, Bf,
                                                       (short)0, acc[h],
                                                       false, false);
    }
  }

  // ---- store C (16x16 f32 layout: row = v + 8*hi, col = ml) --------------
#pragma unroll
  for (int v = 0; v < 8; ++v) {
    int oc = m0 + wm * 16 + hi * 8 + v;
    if (oc >= OC) continue;
#pragma unroll
    for (int h = 0; h < 4; ++h) {
      int nn = n0 + wn * 64 + h * 16 + ml;
      if (nn >= Ntot) continue;
      float val = acc[h][v];
      if (ep == 1) val = tanhf(val + bias[oc]);
      int b = nn / OHW, r = nn - b * OHW;
      int oy = r / OW, ox = r - oy * OW;
      Y[((b * yct) + yc0 + oc) * OH * OW + oy * OW + ox] = val;
    }
  }
}

// ---------------------------------------------------------------------------
// BatchNorm (batch stats over (0,2,3)) -> per-channel scale/shift, then apply.
// ---------------------------------------------------------------------------
__global__ __launch_bounds__(256)
void bn_stats(const float* __restrict__ X, float* __restrict__ ss,
              const float* __restrict__ gamma, const float* __restrict__ beta,
              int ct, int c0, int HW, int C)
{
  int c = blockIdx.x;
  float s = 0.0f, s2 = 0.0f;
  int n = NB * HW;
  for (int i = threadIdx.x; i < n; i += 256) {
    int b = i / HW, r = i - b * HW;
    float v = X[((b * ct) + c0 + c) * HW + r];
    s += v;  s2 += v * v;
  }
  __shared__ float red[256], red2[256];
  red[threadIdx.x] = s;  red2[threadIdx.x] = s2;
  __syncthreads();
  for (int o = 128; o > 0; o >>= 1) {
    if (threadIdx.x < o) {
      red[threadIdx.x]  += red[threadIdx.x + o];
      red2[threadIdx.x] += red2[threadIdx.x + o];
    }
    __syncthreads();
  }
  if (threadIdx.x == 0 && c < C) {
    float inv = 1.0f / (float)n;
    float mu  = red[0] * inv;
    float var = red2[0] * inv - mu * mu;
    float sc  = gamma[c] * rsqrtf(var + 1e-5f);
    ss[2 * c]     = sc;
    ss[2 * c + 1] = beta[c] - mu * sc;
  }
}

__global__ void bn_apply(float* __restrict__ X, const float* __restrict__ ss,
                         int C, int ct, int c0, int HW, int total)
{
  int i = blockIdx.x * 256 + threadIdx.x;
  if (i >= total) return;
  int r = i % HW;
  int c = (i / HW) % C;
  int b = i / (HW * C);
  int idx = ((b * ct) + c0 + c) * HW + r;
  X[idx] = X[idx] * ss[2 * c] + ss[2 * c + 1];
}

// ---------------------------------------------------------------------------
// Fused "zipping": grouped 3x3 conv over [x(3ch) ++ cond-plane] per group.
// ---------------------------------------------------------------------------
__global__ void zip_kernel(const float* __restrict__ x, const float* __restrict__ cond,
                           const float* __restrict__ zw, float* __restrict__ z,
                           int H, int W)
{
  int idx = blockIdx.x * 256 + threadIdx.x;
  int total = NB * 8 * H * W;
  if (idx >= total) return;
  int ox = idx % W;
  int oy = (idx / W) % H;
  int g  = (idx / (W * H)) & 7;
  int b  = idx / (W * H * 8);
  float cv = cond[b * 8 + g];
  float acc = 0.0f;
  for (int ky = 0; ky < 3; ++ky) {
    int iy = oy + ky - 1;
    if (iy < 0 || iy >= H) continue;
    for (int kx = 0; kx < 3; ++kx) {
      int ix = ox + kx - 1;
      if (ix < 0 || ix >= W) continue;
      const float* wg = zw + ((g * 4) * 3 + ky) * 3 + kx;   // [g][c][ky][kx]
      acc += wg[0]  * x[((b * 3 + 0) * H + iy) * W + ix];
      acc += wg[9]  * x[((b * 3 + 1) * H + iy) * W + ix];
      acc += wg[18] * x[((b * 3 + 2) * H + iy) * W + ix];
      acc += wg[27] * cv;
    }
  }
  z[idx] = acc;
}

// ---------------------------------------------------------------------------
// Input index map (setup_inputs flattening order):
//  0 x, 1 cond, 2 zip_w,
//  3 b0.down_w, 4 b0.up_w, 5 b0.up_bias,
//  6.. b1..b6: down_w, dn_g, dn_b, up_w, up_g, up_b (6 each)
//  42 b7.down_w, 43 b7.up_w, 44 b7.up_g, 45 b7.up_b
// ---------------------------------------------------------------------------
extern "C" void kernel_launch(void* const* d_in, const int* in_sizes, int n_in,
                              void* d_out, int out_size, void* d_ws, size_t ws_size,
                              hipStream_t stream)
{
  const float* x    = (const float*)d_in[0];
  const float* cond = (const float*)d_in[1];
  const float* zw   = (const float*)d_in[2];

  float* ws = (float*)d_ws;
  size_t off = 0;
  float* Z  = ws + off; off += (size_t)NB * 8    * 256 * 256;
  float* C1 = ws + off; off += (size_t)NB * 128  * 128 * 128;  // [x1 | h1]
  float* C2 = ws + off; off += (size_t)NB * 256  * 64  * 64;   // [x2 | h2]
  float* C3 = ws + off; off += (size_t)NB * 512  * 32  * 32;
  float* C4 = ws + off; off += (size_t)NB * 1024 * 16  * 16;
  float* C5 = ws + off; off += (size_t)NB * 1024 * 8   * 8;
  float* C6 = ws + off; off += (size_t)NB * 1024 * 4   * 4;
  float* C7 = ws + off; off += (size_t)NB * 1024 * 2   * 2;
  float* X8 = ws + off; off += (size_t)NB * 512;               // bottleneck 1x1
  float* ST = ws + off; off += 2048;                           // bn scale/shift

  {
    int total = NB * 8 * 256 * 256;
    zip_kernel<<<(total + 255) / 256, 256, 0, stream>>>(x, cond, zw, Z, 256, 256);
  }

  auto conv = [&](int widx, const float* Xp, float* Yp, const float* biasp,
                  int OC, int IC, int IH, int OHd,
                  int xct, int xc0, int yct, int yc0, int act, int tr, int ep) {
    int Ntot = NB * OHd * OHd;
    dim3 grid((Ntot + BN - 1) / BN, (OC + BM - 1) / BM);
    conv_wmma<<<grid, 256, 0, stream>>>((const float*)d_in[widx], Xp, Yp, biasp,
        OC, IC, IH, IH, OHd, OHd, xct, xc0, yct, yc0, act, tr, ep);
  };
  auto bn = [&](float* Xp, int gidx, int bidx, int C, int ct, int c0, int S) {
    bn_stats<<<C, 256, 0, stream>>>(Xp, ST, (const float*)d_in[gidx],
                                    (const float*)d_in[bidx], ct, c0, S * S, C);
    int total = NB * C * S * S;
    bn_apply<<<(total + 255) / 256, 256, 0, stream>>>(Xp, ST, C, ct, c0, S * S, total);
  };

  // ---- down path (conv k4 s2 p1; leaky pre-act except outermost) ----------
  conv(3,  Z,  C1, nullptr,  64,   8, 256, 128,    8, 0,  128, 0, 0, 0, 0);
  conv(6,  C1, C2, nullptr, 128,  64, 128,  64,  128, 0,  256, 0, 1, 0, 0);  bn(C2,  7,  8, 128,  256, 0, 64);
  conv(12, C2, C3, nullptr, 256, 128,  64,  32,  256, 0,  512, 0, 1, 0, 0);  bn(C3, 13, 14, 256,  512, 0, 32);
  conv(18, C3, C4, nullptr, 512, 256,  32,  16,  512, 0, 1024, 0, 1, 0, 0);  bn(C4, 19, 20, 512, 1024, 0, 16);
  conv(24, C4, C5, nullptr, 512, 512,  16,   8, 1024, 0, 1024, 0, 1, 0, 0);  bn(C5, 25, 26, 512, 1024, 0,  8);
  conv(30, C5, C6, nullptr, 512, 512,   8,   4, 1024, 0, 1024, 0, 1, 0, 0);  bn(C6, 31, 32, 512, 1024, 0,  4);
  conv(36, C6, C7, nullptr, 512, 512,   4,   2, 1024, 0, 1024, 0, 1, 0, 0);  bn(C7, 37, 38, 512, 1024, 0,  2);
  conv(42, C7, X8, nullptr, 512, 512,   2,   1, 1024, 0,  512, 0, 1, 0, 0);  // innermost (no BN)

  // ---- up path (convT k4 s2 p1; relu pre-act; write into concat 2nd half) -
  conv(43, X8, C7, nullptr, 512,  512,  1,   2,  512, 0, 1024, 512, 2, 1, 0);  bn(C7, 44, 45, 512, 1024, 512,  2);
  conv(39, C7, C6, nullptr, 512, 1024,  2,   4, 1024, 0, 1024, 512, 2, 1, 0);  bn(C6, 40, 41, 512, 1024, 512,  4);
  conv(33, C6, C5, nullptr, 512, 1024,  4,   8, 1024, 0, 1024, 512, 2, 1, 0);  bn(C5, 34, 35, 512, 1024, 512,  8);
  conv(27, C5, C4, nullptr, 512, 1024,  8,  16, 1024, 0, 1024, 512, 2, 1, 0);  bn(C4, 28, 29, 512, 1024, 512, 16);
  conv(21, C4, C3, nullptr, 256, 1024, 16,  32, 1024, 0,  512, 256, 2, 1, 0);  bn(C3, 22, 23, 256,  512, 256, 32);
  conv(15, C3, C2, nullptr, 128,  512, 32,  64,  512, 0,  256, 128, 2, 1, 0);  bn(C2, 16, 17, 128,  256, 128, 64);
  conv(9,  C2, C1, nullptr,  64,  256, 64, 128,  256, 0,  128,  64, 2, 1, 0);  bn(C1, 10, 11,  64,  128,  64, 128);
  // outermost up: relu -> convT -> +bias -> tanh, straight into d_out
  conv(4,  C1, (float*)d_out, (const float*)d_in[5],
           3, 128, 128, 256, 128, 0, 3, 0, 2, 1, 1);

  (void)in_sizes; (void)n_in; (void)out_size; (void)ws_size;
}
